// DisjointSet_37744172597259
// MI455X (gfx1250) — compile-verified, matrix operand order checked by reading, not run on hardware
//
#include <hip/hip_runtime.h>
#include <hip/hip_bf16.h>
#include <math.h>

typedef __attribute__((ext_vector_type(16))) __bf16 v16bf;
typedef __attribute__((ext_vector_type(8)))  __bf16 v8bf;
typedef __attribute__((ext_vector_type(4)))  __bf16 v4bf;
typedef __attribute__((ext_vector_type(8)))  float  v8f;
typedef __attribute__((ext_vector_type(4)))  unsigned int u32x4;
typedef __attribute__((ext_vector_type(8)))  int    i32x8;
typedef __attribute__((ext_vector_type(4)))  int    i32x4;

#define D_      2048
#define H_      128
#define E_      8192
#define NNODES_ 100000
#define SIMTH_  0.7f
#define EPS_    1e-8f

// ---------------------------------------------------------------------------
// Kernel A0: one-time W1 f32 -> bf16 conversion (1 MB in workspace).
// ---------------------------------------------------------------------------
__global__ __launch_bounds__(256) void w1_to_bf16_kernel(
    const float* __restrict__ W1, __bf16* __restrict__ w1bf)
{
    int base = (blockIdx.x * 256 + threadIdx.x) * 4;
    if (base < H_ * 2 * D_) {
        float4 v = *(const float4*)(W1 + base);
        v4bf o;
        o[0] = (__bf16)v.x; o[1] = (__bf16)v.y;
        o[2] = (__bf16)v.z; o[3] = (__bf16)v.w;
        *(v4bf*)(w1bf + base) = o;
    }
}

// ---------------------------------------------------------------------------
// TDM helper: DMA a 128-row x 32-col bf16 tile (row stride 4096 elements)
// from global into LDS, with hardware padding: every 16 DWORDs (32 bf16) of
// payload, insert 4 DWORDs (8 bf16) -> LDS row stride of 40 bf16 elements,
// matching the padded LDS arrays below (bank-conflict-free B reads).
// D# built per CDNA5 ISA ch.8 (group0/group1; groups 2/3 unused for 2D).
// ---------------------------------------------------------------------------
__device__ inline void tdm_load_w1_tile(unsigned lds_byte_off, const __bf16* gtile)
{
    unsigned long long ga = (unsigned long long)(size_t)gtile;

    u32x4 g0;
    g0[0] = 1u;                                   // count=1, user descriptor
    g0[1] = lds_byte_off;                         // lds_addr (bytes)
    g0[2] = (unsigned)(ga & 0xFFFFFFFFu);         // global_addr[31:0]
    g0[3] = (unsigned)((ga >> 32) & 0x01FFFFFFu)  // global_addr[56:32]
          | (2u << 30);                           // type = 2 ("image")

    i32x8 g1;
    g1[0] = (int)((1u << 16)      // data_size = 1 -> 2-byte elements
                | (1u << 20)      // pad_enable
                | (3u << 22)      // pad_interval: 16 DWORDs (64B) of payload
                | (3u << 25));    // pad_amount:   4 DWORDs (16B) of pad
    g1[1] = (int)(4096u << 16);   // tensor_dim0 = 4096 (lo16 in [31:16])
    g1[2] = (int)(128u  << 16);   // tensor_dim0 hi = 0; tensor_dim1 = 128 (lo16)
    g1[3] = (int)(32u   << 16);   // tensor_dim1 hi = 0; tile_dim0 = 32
    g1[4] = (int)128;             // tile_dim1 = 128; tile_dim2 = 0
    g1[5] = (int)4096;            // tensor_dim0_stride = 4096 (lo32)
    g1[6] = 0;                    // stride hi; tensor_dim1_stride lo (unused, 2D)
    g1[7] = 0;

    i32x4 gz;
    gz[0] = 0; gz[1] = 0; gz[2] = 0; gz[3] = 0;   // groups 2/3: unused (2D tile)

#if defined(__clang_major__) && (__clang_major__ >= 23)
    i32x8 gz8;
    gz8[0] = 0; gz8[1] = 0; gz8[2] = 0; gz8[3] = 0;
    gz8[4] = 0; gz8[5] = 0; gz8[6] = 0; gz8[7] = 0;
    __builtin_amdgcn_tensor_load_to_lds(g0, g1, gz, gz, gz8, 0);
#else
    __builtin_amdgcn_tensor_load_to_lds(g0, g1, gz, gz, 0);
#endif
}

// ---------------------------------------------------------------------------
// Kernel A: fused  h = relu([x|y] @ W1^T + b1)  +  cosine-sim reductions.
// 64 blocks x 256 threads (8 wave32). Each wave: one 16-row M tile, all 8
// 16-col N tiles of H=128. Per 32-wide K-step: W1 bf16 tiles arrive via TDM
// (tensor_load_to_lds, TENSORcnt), then 16 v_wmma_f32_16x16x32_bf16.
// ---------------------------------------------------------------------------
__global__ __launch_bounds__(256) void gate_gemm_kernel(
    const float* __restrict__ xf, const float* __restrict__ yf,
    const __bf16* __restrict__ w1bf, const float* __restrict__ b1,
    float* __restrict__ h, float* __restrict__ sim)
{
    // 40-element rows (80 B): produced by TDM padding; 20-bank stride -> no
    // bank conflicts across the 16 lanes of a B-fragment read.
    __shared__ __attribute__((aligned(16))) __bf16 ldsWx[H_][40];
    __shared__ __attribute__((aligned(16))) __bf16 ldsWy[H_][40];

    const int tid  = threadIdx.x;
    const int lane = tid & 31;
    const int wave = tid >> 5;
    const int c    = lane & 15;     // A-row-in-tile / B-col-in-tile
    const int hsel = lane >> 4;     // half select (ISA 16-bit A/B lane layout)
    const int kb   = hsel * 8;      // K sub-base inside a 32-wide step

    const int mbase = (blockIdx.x * 8 + wave) * 16;
    const int row   = mbase + c;
    const float* __restrict__ xrow = xf + (size_t)row * D_;
    const float* __restrict__ yrow = yf + (size_t)row * D_;

    v8f acc[8];
#pragma unroll
    for (int t = 0; t < 8; ++t) acc[t] = (v8f)0.0f;

    float dot = 0.f, nx2 = 0.f, ny2 = 0.f;

    const unsigned ldsx_off = (unsigned)(size_t)(void*)&ldsWx[0][0];
    const unsigned ldsy_off = (unsigned)(size_t)(void*)&ldsWy[0][0];

    for (int k0 = 0; k0 < D_; k0 += 32) {
        // ---- TDM: DMA W1 x-half / y-half tiles into LDS (wave 0 issues) ----
        if (wave == 0) {
            tdm_load_w1_tile(ldsx_off, w1bf + k0);        // cols [k0, k0+32)
            tdm_load_w1_tile(ldsy_off, w1bf + D_ + k0);   // cols [2048+k0, ..)
            __builtin_amdgcn_s_wait_tensorcnt(0);
        }
        __syncthreads();

        // prefetch next K-step of this lane's A rows (global_prefetch_b8)
        if (k0 + 32 < D_) {
            __builtin_prefetch(xrow + k0 + 32 + kb, 0, 3);
            __builtin_prefetch(yrow + k0 + 32 + kb, 0, 3);
        }

        // ---- A fragments (ISA 16-bit A layout) + fused sim reductions ----
        const float* xp = xrow + k0 + kb;
        const float* yp = yrow + k0 + kb;
        float4 x0 = ((const float4*)xp)[0];
        float4 x1 = ((const float4*)xp)[1];
        float4 x2 = ((const float4*)(xp + 16))[0];
        float4 x3 = ((const float4*)(xp + 16))[1];
        float4 y0 = ((const float4*)yp)[0];
        float4 y1 = ((const float4*)yp)[1];
        float4 y2 = ((const float4*)(yp + 16))[0];
        float4 y3 = ((const float4*)(yp + 16))[1];

        float xv[16] = {x0.x, x0.y, x0.z, x0.w, x1.x, x1.y, x1.z, x1.w,
                        x2.x, x2.y, x2.z, x2.w, x3.x, x3.y, x3.z, x3.w};
        float yv[16] = {y0.x, y0.y, y0.z, y0.w, y1.x, y1.y, y1.z, y1.w,
                        y2.x, y2.y, y2.z, y2.w, y3.x, y3.y, y3.z, y3.w};

        v16bf ax, ay;
#pragma unroll
        for (int i = 0; i < 16; ++i) {
            dot += xv[i] * yv[i];
            nx2 += xv[i] * xv[i];
            ny2 += yv[i] * yv[i];
            ax[i] = (__bf16)xv[i];
            ay[i] = (__bf16)yv[i];
        }

        // ---- 8 N-tiles: B fragments from LDS, 2 WMMAs each (x + y halves) ----
#pragma unroll
        for (int t = 0; t < 8; ++t) {
            const __bf16* px = &ldsWx[t * 16 + c][kb];
            const __bf16* py = &ldsWy[t * 16 + c][kb];
            v8bf bxl = *(const v8bf*)px;
            v8bf bxh = *(const v8bf*)(px + 16);
            v8bf byl = *(const v8bf*)py;
            v8bf byh = *(const v8bf*)(py + 16);
            v16bf bx, by;
#pragma unroll
            for (int i = 0; i < 8; ++i) {
                bx[i] = bxl[i]; bx[8 + i] = bxh[i];
                by[i] = byl[i]; by[8 + i] = byh[i];
            }
            acc[t] = __builtin_amdgcn_wmma_f32_16x16x32_bf16(
                false, ax, false, bx, (short)0, acc[t], false, false);
            acc[t] = __builtin_amdgcn_wmma_f32_16x16x32_bf16(
                false, ay, false, by, (short)0, acc[t], false, false);
        }
        __syncthreads();
    }

    // ---- cosine sim: lanes (m) and (m+16) jointly cover all K ----
    dot += __shfl_xor(dot, 16, 32);
    nx2 += __shfl_xor(nx2, 16, 32);
    ny2 += __shfl_xor(ny2, 16, 32);
    if (hsel == 0) {
        float nx = fmaxf(sqrtf(nx2), EPS_);
        float ny = fmaxf(sqrtf(ny2), EPS_);
        sim[row] = dot / (nx * ny);
    }

    // ---- epilogue: +b1, relu, store h (ISA f32 C/D layout) ----
#pragma unroll
    for (int t = 0; t < 8; ++t) {
        int n = t * 16 + c;
        float bias = b1[n];
#pragma unroll
        for (int r = 0; r < 8; ++r) {
            int m = mbase + r + 8 * hsel;
            float v = acc[t][r] + bias;
            h[(size_t)m * H_ + n] = v > 0.f ? v : 0.f;
        }
    }
}

// ---------------------------------------------------------------------------
// Kernel B: attn = sigmoid(h @ W2^T + b2); w = (sim >= 0.7) ? attn : 0
// ---------------------------------------------------------------------------
__global__ __launch_bounds__(256) void attn_kernel(
    const float* __restrict__ h, const float* __restrict__ W2,
    const float* __restrict__ b2, const float* __restrict__ sim,
    float* __restrict__ w_out, int* __restrict__ mask_out)
{
    int e = blockIdx.x * 256 + threadIdx.x;
    if (e >= E_) return;
    const float4* hp = (const float4*)(h + (size_t)e * H_);
    const float4* wp = (const float4*)W2;
    float z = b2[0];
#pragma unroll 4
    for (int i = 0; i < H_ / 4; ++i) {
        float4 hv = hp[i];
        float4 wv = wp[i];
        z += hv.x * wv.x + hv.y * wv.y + hv.z * wv.z + hv.w * wv.w;
    }
    float attn = 1.0f / (1.0f + __expf(-z));
    int m = sim[e] >= SIMTH_ ? 1 : 0;
    w_out[e]    = m ? attn : 0.0f;
    mask_out[e] = m;
}

// ---------------------------------------------------------------------------
// Kernel C: copy parent/rank into the output buffers (inputs stay pristine)
// ---------------------------------------------------------------------------
__global__ __launch_bounds__(256) void copy_state_kernel(
    const int* __restrict__ parent_in, const float* __restrict__ rank_in,
    int* __restrict__ parent_out, float* __restrict__ rank_out)
{
    int i = blockIdx.x * 256 + threadIdx.x;
    if (i < NNODES_) {
        parent_out[i] = parent_in[i];
        rank_out[i]   = rank_in[i];
    }
}

// ---------------------------------------------------------------------------
// Kernel D: sequential union-find with path compression (matches lax.scan)
// ---------------------------------------------------------------------------
__global__ void uf_kernel(
    const int* __restrict__ x_idx, const int* __restrict__ y_idx,
    const float* __restrict__ w, const int* __restrict__ mask,
    int* parent, float* rank)
{
    if (threadIdx.x != 0 || blockIdx.x != 0) return;
    for (int e = 0; e < E_; ++e) {
        if (!mask[e]) continue;
        int x = x_idx[e], y = y_idx[e];

        int rx = x;
        while (parent[rx] != rx) rx = parent[rx];
        int j = x;
        while (j != rx) { int nj = parent[j]; parent[j] = rx; j = nj; }

        int ry = y;
        while (parent[ry] != ry) ry = parent[ry];
        j = y;
        while (j != ry) { int nj = parent[j]; parent[j] = ry; j = nj; }

        if (rx == ry) continue;  // same root: no structural / rank change
        bool bigger_x = rank[rx] > rank[ry];
        int big   = bigger_x ? rx : ry;
        int small = bigger_x ? ry : rx;
        parent[small] = big;
        rank[big] += rank[small] * w[e];
    }
}

// ---------------------------------------------------------------------------
extern "C" void kernel_launch(void* const* d_in, const int* in_sizes, int n_in,
                              void* d_out, int out_size, void* d_ws, size_t ws_size,
                              hipStream_t stream)
{
    const int*   x_idx     = (const int*)d_in[0];
    const int*   y_idx     = (const int*)d_in[1];
    const float* x_feat    = (const float*)d_in[2];
    const float* y_feat    = (const float*)d_in[3];
    const float* W1        = (const float*)d_in[4];
    const float* b1        = (const float*)d_in[5];
    const float* W2        = (const float*)d_in[6];
    const float* b2        = (const float*)d_in[7];
    const int*   parent_in = (const int*)d_in[8];
    const float* rank_in   = (const float*)d_in[9];

    float* out        = (float*)d_out;
    float* w_out      = out;                          // [E]
    int*   parent_out = (int*)(out + E_);             // [N] (int bits)
    float* rank_out   = out + E_ + NNODES_;           // [N]

    float*  ws   = (float*)d_ws;
    float*  sim  = ws;                                       // [E]
    float*  h    = ws + E_;                                  // [E*H]
    int*    mask = (int*)(ws + E_ + (size_t)E_ * H_);        // [E]
    __bf16* w1bf = (__bf16*)(ws + E_ + (size_t)E_ * H_ + E_);// [H * 2D]

    w1_to_bf16_kernel<<<(H_ * 2 * D_ / 4 + 255) / 256, 256, 0, stream>>>(W1, w1bf);
    gate_gemm_kernel<<<E_ / 128, 256, 0, stream>>>(x_feat, y_feat, w1bf, b1, h, sim);
    attn_kernel<<<E_ / 256, 256, 0, stream>>>(h, W2, b2, sim, w_out, mask);
    copy_state_kernel<<<(NNODES_ + 255) / 256, 256, 0, stream>>>(parent_in, rank_in,
                                                                 parent_out, rank_out);
    uf_kernel<<<1, 32, 0, stream>>>(x_idx, y_idx, w_out, mask, parent_out, rank_out);
}
